// TextLSTM_byMyself_50611894616346
// MI455X (gfx1250) — matrix operand order repaired and checked by reading, not compiled, not run on hardware
//
#include <hip/hip_runtime.h>
#include <hip/hip_fp16.h>

// ---------------------------------------------------------------------------
// 2-layer LSTM LM on gfx1250 (MI455X), fp16 WMMA + CDNA5 async data movement.
//   - persistent recurrence kernel: 8 WGs x 256 threads (wave32), batch tile
//     of 32 rows; x/h/c state resident in LDS for all 64 steps; h double-
//     buffered by parity; LSTM cell fused on WMMA D-fragments.
//   - fp16 weights + fp16 embedding table pre-converted into d_ws; recurrent
//     weights (3.25 MB) stream from L2 (192 MB) every step.
//   - embedding gather per step: GLOBAL_LOAD_ASYNC_TO_LDS_B128 (ASYNCcnt).
//   - logits kernel stages its h tile with the Tensor Data Mover
//     (tensor_load_to_lds, hardware LDS row padding, TENSORcnt).
// ---------------------------------------------------------------------------

#define N_CLASS 32000
#define EMB     256
#define HID     512
#define BATCH   256
#define SEQ     64
#define BTILE   32          // batch rows per workgroup (2 WMMA M-tiles)
#define NWAVE   8
#define XSTR    (EMB + 8)   // 264 f16 -> 528 B rows (16B aligned)
#define HSTR    (HID + 8)   // 520 f16 -> 1040 B rows (16B aligned)

typedef __attribute__((ext_vector_type(16))) _Float16     v16h;
typedef __attribute__((ext_vector_type(8)))  float        v8f;
typedef __attribute__((ext_vector_type(4)))  unsigned int u32x4;
typedef __attribute__((ext_vector_type(8)))  unsigned int u32x8;

union F16x16 { v16h v; float4 q[2]; _Float16 h[16]; };

// A-fragment (16x32 f16) from an LDS matrix with row stride `stride` (f16
// units, multiple of 8). ISA layout: lane holds row M=(lane&15)+mrow0;
// elements 0..7 -> K = kbase + 8*(lane>>4) + e ; elements 8..15 -> +16.
__device__ __forceinline__ v16h a_frag_lds(const _Float16* base, int stride,
                                           int mrow0, int kbase, int lane) {
  const int hw = lane >> 4;
  const _Float16* p = base + (size_t)(mrow0 + (lane & 15)) * stride + kbase + hw * 8;
  F16x16 u;
  u.q[0] = *(const float4*)(p);
  u.q[1] = *(const float4*)(p + 16);
  return u.v;
}

// B-fragment (32x16 f16) from a global [out,K] row-major fp16 matrix.
// ISA layout: lane holds column n=(lane&15); elements 0..7 -> K = kbase + 8*(lane>>4) + e ; elements 8..15 -> +16.
__device__ __forceinline__ v16h b_frag_g(const _Float16* __restrict__ w,
                                         int row, int K, int kbase, int lane) {
  const int hw = lane >> 4;
  const _Float16* p = w + (size_t)row * K + kbase + hw * 8;
  F16x16 u;
  u.q[0] = *(const float4*)(p);
  u.q[1] = *(const float4*)(p + 16);
  return u.v;
}

__device__ __forceinline__ v8f wmma16(v16h a, v16h b, v8f c) {
  // emits v_wmma_f32_16x16x32_f16
  return __builtin_amdgcn_wmma_f32_16x16x32_f16(false, a, false, b, (short)0, c,
                                                false, false);
}

__device__ __forceinline__ float sigm(float x)   { return 1.0f / (1.0f + __expf(-x)); }
__device__ __forceinline__ float tanh_f(float x) { return 2.0f * sigm(2.0f * x) - 1.0f; }

// low 32 bits of a generic pointer to LDS == byte offset in the wave's LDS
__device__ __forceinline__ unsigned lds_off_of(const void* p) {
  return (unsigned)(unsigned long long)p;
}

// per-lane async 16B copy global -> LDS (ASYNCcnt)
__device__ __forceinline__ void async_ld_b128(unsigned lds_byte_off, const void* g) {
  asm volatile("global_load_async_to_lds_b128 %0, %1, off"
               :: "v"(lds_byte_off), "v"(g) : "memory");
}
__device__ __forceinline__ void wait_asynccnt0() {
  asm volatile("s_wait_asynccnt 0x0" ::: "memory");
}

// Tensor Data Mover: 2D tile load (groups 2/3 = NULL)
__device__ __forceinline__ void tdm_load_2d(u32x4 g0, u32x8 g1) {
  asm volatile("tensor_load_to_lds %0, %1" :: "s"(g0), "s"(g1) : "memory");
}

// ---------------------------------------------------------------------------
// fp32 -> fp16 conversion (grid-stride)
// ---------------------------------------------------------------------------
__global__ void lstm_cvt_f16(const float* __restrict__ s, _Float16* __restrict__ d, int n) {
  int i = blockIdx.x * blockDim.x + threadIdx.x;
  int stride = gridDim.x * blockDim.x;
  for (; i < n; i += stride) d[i] = (_Float16)s[i];
}

// ---------------------------------------------------------------------------
// Persistent recurrence kernel. grid = BATCH/BTILE blocks of 256 threads.
// wx : [4*HID][EMB] f16 (gate order i,f,g,o)   wh : [4*HID][HID] f16
// wp : [EMB][HID]  f16 (h->x projection)       cf16: [N_CLASS][EMB] f16
// hfin: [BATCH][HID] f16 out
// ---------------------------------------------------------------------------
__global__ void lstm_rec_wmma(const int* __restrict__ X, const _Float16* __restrict__ cf16,
                              const _Float16* __restrict__ wx, const _Float16* __restrict__ wh,
                              const _Float16* __restrict__ wp,
                              const float* __restrict__ bi, const float* __restrict__ bff,
                              const float* __restrict__ bg, const float* __restrict__ bo,
                              const float* __restrict__ bp, _Float16* __restrict__ hfin) {
  extern __shared__ __align__(16) char smem[];
  _Float16* xin  = (_Float16*)smem;                   // [BTILE][XSTR]
  _Float16* hbuf = xin + BTILE * XSTR;                // [2 layer][2 parity][BTILE][HSTR]
  float*    cbuf = (float*)(hbuf + 4 * BTILE * HSTR); // [2 layer][BTILE][HID]

  const int lane = threadIdx.x & 31;
  const int wid  = threadIdx.x >> 5;
  const int hw   = lane >> 4;
  const int nn   = lane & 15;
  const int row0 = blockIdx.x * BTILE;
  const unsigned xin_lds = lds_off_of(xin);

  // zero initial state (h0 = c0 = 0)
  for (int i = threadIdx.x; i < 4 * BTILE * HSTR; i += blockDim.x) hbuf[i] = (_Float16)0.0f;
  for (int i = threadIdx.x; i < 2 * BTILE * HID;  i += blockDim.x) cbuf[i] = 0.0f;
  __syncthreads();

  for (int t = 0; t < SEQ; ++t) {
    // ---- async gather of C16[X[:,t]] -> xin (4 x 16B per thread) ----
    {
      const int b  = threadIdx.x >> 3;                 // batch row 0..31
      const int c0 = (threadIdx.x & 7) * 4;            // 16B-chunk base 0..28
      const int idx = X[(row0 + b) * SEQ + t];
      const _Float16* src = cf16 + (size_t)idx * EMB;
      const unsigned dst = xin_lds + (unsigned)b * (XSTR * 2);
      #pragma unroll
      for (int i = 0; i < 4; ++i)
        async_ld_b128(dst + (unsigned)(c0 + i) * 16u, src + (c0 + i) * 8);
      wait_asynccnt0();
    }
    __syncthreads();

    const int rd = t & 1;
    const int wr = 1 - rd;

    for (int l = 0; l < 2; ++l) {
      const _Float16* hr  = hbuf + (l * 2 + rd) * BTILE * HSTR;
      _Float16*       hwb = hbuf + (l * 2 + wr) * BTILE * HSTR;
      float*          cl  = cbuf + l * BTILE * HID;

      // ---- gate GEMMs: each wave owns 4 hidden-unit tiles of 16 ----
      for (int jt = 0; jt < 4; ++jt) {
        const int j = (wid * 4 + jt) * 16;
        v8f acc[4][2];
        {
          const float binit[4] = { bi[j + nn], bff[j + nn], bg[j + nn], bo[j + nn] };
          #pragma unroll
          for (int g = 0; g < 4; ++g) {
            v8f z;
            #pragma unroll
            for (int r = 0; r < 8; ++r) z[r] = binit[g];
            acc[g][0] = z; acc[g][1] = z;
          }
        }
        // input part: K = EMB
        for (int k = 0; k < EMB; k += 32) {
          v16h a0 = a_frag_lds(xin, XSTR, 0,  k, lane);
          v16h a1 = a_frag_lds(xin, XSTR, 16, k, lane);
          #pragma unroll
          for (int g = 0; g < 4; ++g) {
            v16h bfr = b_frag_g(wx, g * HID + j + nn, EMB, k, lane);
            acc[g][0] = wmma16(a0, bfr, acc[g][0]);
            acc[g][1] = wmma16(a1, bfr, acc[g][1]);
          }
        }
        // recurrent part: K = HID
        for (int k = 0; k < HID; k += 32) {
          v16h a0 = a_frag_lds(hr, HSTR, 0,  k, lane);
          v16h a1 = a_frag_lds(hr, HSTR, 16, k, lane);
          #pragma unroll
          for (int g = 0; g < 4; ++g) {
            v16h bfr = b_frag_g(wh, g * HID + j + nn, HID, k, lane);
            acc[g][0] = wmma16(a0, bfr, acc[g][0]);
            acc[g][1] = wmma16(a1, bfr, acc[g][1]);
          }
        }
        // ---- fused LSTM cell on D-fragments (m = 16*mt + 8*hw + r, n = nn) ----
        #pragma unroll
        for (int m = 0; m < 2; ++m) {
          #pragma unroll
          for (int r = 0; r < 8; ++r) {
            const int rowl = m * 16 + hw * 8 + r;
            const int col  = j + nn;
            const float ig = sigm(acc[0][m][r]);
            const float fg = sigm(acc[1][m][r]);
            const float gg = tanh_f(acc[2][m][r]);
            const float og = sigm(acc[3][m][r]);
            const float cn = fg * cl[rowl * HID + col] + ig * gg;
            cl[rowl * HID + col] = cn;                       // c owned by this lane only
            hwb[rowl * HSTR + col] = (_Float16)(og * tanh_f(cn));
          }
        }
      }
      __syncthreads();

      // ---- projection x = h0_new @ Wp^T + b (layer-1 projection is unused) ----
      if (l == 0) {
        const _Float16* hsrc = hbuf + (0 * 2 + wr) * BTILE * HSTR;
        for (int ntl = 0; ntl < 2; ++ntl) {
          const int ncol = (wid * 2 + ntl) * 16;
          v8f p0, p1;
          {
            const float bv = bp[ncol + nn];
            #pragma unroll
            for (int r = 0; r < 8; ++r) { p0[r] = bv; p1[r] = bv; }
          }
          for (int k = 0; k < HID; k += 32) {
            v16h a0  = a_frag_lds(hsrc, HSTR, 0,  k, lane);
            v16h a1  = a_frag_lds(hsrc, HSTR, 16, k, lane);
            v16h bfr = b_frag_g(wp, ncol + nn, HID, k, lane);
            p0 = wmma16(a0, bfr, p0);
            p1 = wmma16(a1, bfr, p1);
          }
          #pragma unroll
          for (int r = 0; r < 8; ++r) {
            xin[(0  + hw * 8 + r) * XSTR + ncol + nn] = (_Float16)p0[r];
            xin[(16 + hw * 8 + r) * XSTR + ncol + nn] = (_Float16)p1[r];
          }
        }
        __syncthreads();
      }
    }
  }

  // ---- export final h of layer 1 (parity = SEQ&1 = 0 for even SEQ) ----
  {
    const _Float16* hf = hbuf + (1 * 2 + (SEQ & 1)) * BTILE * HSTR;
    for (int i = threadIdx.x; i < BTILE * HID; i += blockDim.x) {
      const int b = i / HID, k = i % HID;
      hfin[(size_t)(row0 + b) * HID + k] = hf[b * HSTR + k];
    }
  }
}

// ---------------------------------------------------------------------------
// logits = h_fin @ W_final^T + b_final.  grid = (N_CLASS/16/NWAVE, BATCH/BTILE)
// h tile staged into LDS by the Tensor Data Mover with hardware row padding.
// ---------------------------------------------------------------------------
__global__ void lstm_logits_wmma(const _Float16* __restrict__ hfin,
                                 const _Float16* __restrict__ wf,
                                 const float* __restrict__ bfin,
                                 float* __restrict__ out) {
  extern __shared__ __align__(16) char smem[];
  _Float16* hl = (_Float16*)smem;                 // [BTILE][HSTR]
  const int lane = threadIdx.x & 31;
  const int wid  = threadIdx.x >> 5;
  const int hw   = lane >> 4;
  const int nn   = lane & 15;
  const int row0 = blockIdx.y * BTILE;
  const int nt   = blockIdx.x * NWAVE + wid;      // N-tile 0..1999

  if (wid == 0) {
    // D# group 0: count=1 | lds_addr | global_addr(57b) | type=2
    const unsigned long long ga = (unsigned long long)(const void*)(hfin + (size_t)row0 * HID);
    u32x4 g0;
    g0.x = 1u;                                            // count=1, user mode, no gather
    g0.y = lds_off_of(hl);                                // LDS byte address of tile
    g0.z = (unsigned)(ga & 0xffffffffull);                // global_addr[31:0]
    g0.w = (unsigned)((ga >> 32) & 0x1ffffffull) | (2u << 30); // addr[56:32] | type=2
    // D# group 1: 2B elems, pad 1024B rows by 16B -> HSTR layout
    u32x8 g1;
    g1.s0 = (1u << 16)            // data_size = 1 (2 bytes)
          | (1u << 20)            // pad_enable
          | (7u << 22)            // pad_interval: 256 dwords = 1024 B (one row)
          | (3u << 25);           // pad_amount: 4 dwords = 16 B
    g1.s1 = (unsigned)(HID & 0xffff) << 16;   // tensor_dim0[15:0] = 512
    g1.s2 = (unsigned)(BATCH & 0xffff) << 16; // tensor_dim0 hi=0 | tensor_dim1[15:0]=256
    g1.s3 = (unsigned)HID << 16;              // tensor_dim1 hi=0 | tile_dim0=512
    g1.s4 = (unsigned)BTILE;                  // tile_dim1=32 | tile_dim2=0
    g1.s5 = (unsigned)HID;                    // tensor_dim0_stride[31:0]=512
    g1.s6 = 0u;                               // stride hi | tensor_dim1_stride lo
    g1.s7 = 0u;
    tdm_load_2d(g0, g1);
    __builtin_amdgcn_s_wait_tensorcnt(0);
  }
  __syncthreads();

  const int ncol = nt * 16;
  v8f acc0, acc1;
  {
    const float bv = bfin[ncol + nn];
    #pragma unroll
    for (int r = 0; r < 8; ++r) { acc0[r] = bv; acc1[r] = bv; }
  }
  for (int k = 0; k < HID; k += 32) {
    v16h a0  = a_frag_lds(hl, HSTR, 0,  k, lane);
    v16h a1  = a_frag_lds(hl, HSTR, 16, k, lane);
    v16h bfr = b_frag_g(wf, ncol + nn, HID, k, lane);
    acc0 = wmma16(a0, bfr, acc0);
    acc1 = wmma16(a1, bfr, acc1);
  }
  #pragma unroll
  for (int r = 0; r < 8; ++r) {
    out[(size_t)(row0 + 0  + hw * 8 + r) * N_CLASS + ncol + nn] = acc0[r];
    out[(size_t)(row0 + 16 + hw * 8 + r) * N_CLASS + ncol + nn] = acc1[r];
  }
}

// ---------------------------------------------------------------------------
extern "C" void kernel_launch(void* const* d_in, const int* in_sizes, int n_in,
                              void* d_out, int out_size, void* d_ws, size_t ws_size,
                              hipStream_t stream) {
  (void)in_sizes; (void)n_in; (void)out_size; (void)ws_size;

  const int*   X    = (const int*)  d_in[0];
  const float* Cemb = (const float*)d_in[1];
  const float* Wii  = (const float*)d_in[2];
  const float* Whi  = (const float*)d_in[3];
  const float* b_i  = (const float*)d_in[4];
  const float* Wif  = (const float*)d_in[5];
  const float* Whf  = (const float*)d_in[6];
  const float* b_f  = (const float*)d_in[7];
  const float* Wig  = (const float*)d_in[8];
  const float* Whg  = (const float*)d_in[9];
  const float* b_g  = (const float*)d_in[10];
  const float* Wio  = (const float*)d_in[11];
  const float* Who  = (const float*)d_in[12];
  const float* b_o  = (const float*)d_in[13];
  const float* Wp   = (const float*)d_in[14];
  const float* b_p  = (const float*)d_in[15];
  const float* Wfin = (const float*)d_in[16];
  const float* bfin = (const float*)d_in[17];

  // workspace layout (fp16 weights + embedding table + final hidden), ~53 MB
  char* ws = (char*)d_ws;
  const size_t WX_N = (size_t)4 * HID * EMB;      // 524288
  const size_t WH_N = (size_t)4 * HID * HID;      // 1048576
  const size_t WP_N = (size_t)EMB * HID;          // 131072
  const size_t WF_N = (size_t)N_CLASS * HID;      // 16384000
  const size_t CE_N = (size_t)N_CLASS * EMB;      // 8192000
  _Float16* wx   = (_Float16*)(ws);
  _Float16* wh   = wx + WX_N;
  _Float16* wpp  = wh + WH_N;
  _Float16* wf   = wpp + WP_N;
  _Float16* hfin = wf + WF_N;                     // BATCH*HID f16
  _Float16* cf16 = hfin + (size_t)BATCH * HID;

  auto cvt = [&](const float* s, _Float16* d, size_t n) {
    int blocks = (int)((n + 255) / 256);
    if (blocks > 4096) blocks = 4096;
    lstm_cvt_f16<<<dim3(blocks), dim3(256), 0, stream>>>(s, d, (int)n);
  };
  // gate order i,f,g,o, each [HID, K] row-major (== torch [out,in])
  cvt(Wii, wx + 0 * (size_t)HID * EMB, (size_t)HID * EMB);
  cvt(Wif, wx + 1 * (size_t)HID * EMB, (size_t)HID * EMB);
  cvt(Wig, wx + 2 * (size_t)HID * EMB, (size_t)HID * EMB);
  cvt(Wio, wx + 3 * (size_t)HID * EMB, (size_t)HID * EMB);
  cvt(Whi, wh + 0 * (size_t)HID * HID, (size_t)HID * HID);
  cvt(Whf, wh + 1 * (size_t)HID * HID, (size_t)HID * HID);
  cvt(Whg, wh + 2 * (size_t)HID * HID, (size_t)HID * HID);
  cvt(Who, wh + 3 * (size_t)HID * HID, (size_t)HID * HID);
  cvt(Wp,  wpp, WP_N);
  cvt(Wfin, wf, WF_N);
  cvt(Cemb, cf16, CE_N);

  // persistent recurrence: 8 WGs x 256 threads, ~274.5 KB dynamic LDS (< 320 KB/WGP)
  const size_t rec_smem = (size_t)BTILE * XSTR * 2     // x tile
                        + (size_t)4 * BTILE * HSTR * 2 // h double-buffered, 2 layers
                        + (size_t)2 * BTILE * HID * 4; // c, 2 layers
  (void)hipFuncSetAttribute(reinterpret_cast<const void*>(&lstm_rec_wmma),
                            hipFuncAttributeMaxDynamicSharedMemorySize, (int)rec_smem);
  lstm_rec_wmma<<<dim3(BATCH / BTILE), dim3(256), rec_smem, stream>>>(
      X, cf16, wx, wh, wpp, b_i, b_f, b_g, b_o, b_p, hfin);

  // wide logits GEMM: 250 x 8 blocks
  const size_t lg_smem = (size_t)BTILE * HSTR * 2;
  lstm_logits_wmma<<<dim3(N_CLASS / 16 / NWAVE, BATCH / BTILE), dim3(256), lg_smem, stream>>>(
      hfin, wf, bfin, (float*)d_out);
}
